// HybridAffinityModel_781684048636
// MI455X (gfx1250) — matrix operand order repaired and checked by reading, not compiled, not run on hardware
//
#include <hip/hip_runtime.h>

// ---------------------------------------------------------------------------
// CDNA5 (gfx1250, wave32) implementation of the HybridAffinityModel forward
// pass.  All dense layers run through v_wmma_f32_16x16x32_bf16 with f32
// accumulation; graph aggregation / pooling use f32 atomics; BatchNorm batch
// stats use an LDS tree reduction.
//
// GEMM: each wave computes a 16x64 output strip (4 WMMA accumulators) so the
// A fragment is loaded once per K-step and reused across 4 B tiles:
// 10 global_load_b128 per 4 v_wmma (vs 16 naively).
//
// MHA simplification: key length == 1 -> softmax == 1 ->
//   mha(q, kv) == (kv @ Wv + bv) @ Wo + bo       (Wq / Wk are dead code)
// ---------------------------------------------------------------------------

typedef __bf16 bf16_t;
typedef __attribute__((ext_vector_type(16))) __bf16 v16bf;
typedef __attribute__((ext_vector_type(8)))  __bf16 v8bf;
typedef __attribute__((ext_vector_type(8)))  float  v8f;

#define HIDDEN 256
#define TPB 256

static inline int nblocks(long long n, int tpb = TPB) {
  long long b = (n + tpb - 1) / tpb;
  if (b > 1048576) b = 1048576;
  if (b < 1) b = 1;
  return (int)b;
}

// --------------------------- elementwise kernels ---------------------------

__global__ void fill_zero_f32(float* __restrict__ p, long long n) {
  long long i = (long long)blockIdx.x * blockDim.x + threadIdx.x;
  long long st = (long long)gridDim.x * blockDim.x;
  for (; i < n; i += st) p[i] = 0.0f;
}

// agg[dst[e]][f] += x[src[e]][f]
__global__ void scatter_add_kernel(const float* __restrict__ x,
                                   const int* __restrict__ src,
                                   const int* __restrict__ dst,
                                   float* __restrict__ agg,
                                   long long E, int D) {
  long long n = E * (long long)D;
  long long t = (long long)blockIdx.x * blockDim.x + threadIdx.x;
  long long st = (long long)gridDim.x * blockDim.x;
  for (; t < n; t += st) {
    long long e = t / D;
    int f = (int)(t - e * D);
    atomicAdd(&agg[(long long)dst[e] * D + f], x[(long long)src[e] * D + f]);
  }
}

// A[i][k] = bf16(x[i][k] + agg[i][k]) for i<M,k<D ; zero padding elsewhere.
__global__ void prep_a_kernel(const float* __restrict__ x,
                              const float* __restrict__ agg,
                              bf16_t* __restrict__ A,
                              int M, int Mpad, int D, int Kpad) {
  long long n = (long long)Mpad * Kpad;
  long long t = (long long)blockIdx.x * blockDim.x + threadIdx.x;
  long long st = (long long)gridDim.x * blockDim.x;
  for (; t < n; t += st) {
    long long i = t / Kpad;
    int k = (int)(t - i * Kpad);
    float v = 0.0f;
    if (i < M && k < D) {
      long long idx = i * (long long)D + k;
      v = x[idx] + (agg ? agg[idx] : 0.0f);
    }
    A[t] = (bf16_t)v;
  }
}

// Wt[n][k] = bf16(W[k][n]) (transpose + K-pad), W row-major [K][N]
__global__ void prep_wt_kernel(const float* __restrict__ W,
                               bf16_t* __restrict__ Wt,
                               int K, int N, int Kpad) {
  long long n = (long long)N * Kpad;
  long long t = (long long)blockIdx.x * blockDim.x + threadIdx.x;
  long long st = (long long)gridDim.x * blockDim.x;
  for (; t < n; t += st) {
    long long nn = t / Kpad;
    int k = (int)(t - nn * Kpad);
    float v = (k < K) ? W[(long long)k * N + nn] : 0.0f;
    Wt[t] = (bf16_t)v;
  }
}

// --------------------------- WMMA GEMM -------------------------------------
// C[M x N] = act(A[M x Kpad] * Wt[N x Kpad]^T + bias)
// grid = (ceil(N/256), M/16), block = 128 (4 waves).
// Each wave owns a 16x64 output strip: 4 accumulators, A fragment loaded once
// per K-step and reused across 4 B tiles (register-level A reuse).
__global__ void __launch_bounds__(128)
wmma_gemm_kernel(const bf16_t* __restrict__ A, int lda,
                 const bf16_t* __restrict__ W, int ldw,
                 const float* __restrict__ bias,
                 float* __restrict__ Cf, bf16_t* __restrict__ Cb,
                 int ldc, int N, int Kpad, int relu) {
  const int lane = threadIdx.x & 31;
  const int wave = threadIdx.x >> 5;
  const int n0 = (blockIdx.x * 4 + wave) * 64;   // 64 cols per wave
  if (n0 >= N) return;                           // wave-uniform: EXEC stays ~0
  const long long m0 = (long long)blockIdx.y * 16;
  const int half = lane >> 4;       // 0: lanes 0-15, 1: lanes 16-31
  const int mrow = lane & 15;       // A row / B col within tile
  const int acol = half * 8;        // A K sub-offset per 16-bit A layout
  const int bk   = half * 8;        // B K sub-offset per 16-bit B layout

  const bf16_t* __restrict__ Arow = A + (m0 + mrow) * (long long)lda;
  const bf16_t* __restrict__ Wrow = W + (long long)(n0 + mrow) * ldw;
  const long long wstep = (long long)16 * ldw;   // 16 output cols per B tile

  v8f acc[4];
#pragma unroll
  for (int j = 0; j < 4; ++j) acc[j] = (v8f){0.f,0.f,0.f,0.f,0.f,0.f,0.f,0.f};

  for (int k0 = 0; k0 < Kpad; k0 += 32) {
    union { v16bf v; v8bf h[2]; } a;
    a.h[0] = *(const v8bf*)(Arow + k0 + acol);        // K = k0+acol .. +7
    a.h[1] = *(const v8bf*)(Arow + k0 + 16 + acol);   // K = k0+16+acol .. +7
#pragma unroll
    for (int j = 0; j < 4; ++j) {
      union { v16bf v; v8bf h[2]; } bu;
      bu.h[0] = *(const v8bf*)(Wrow + j * wstep + k0 + bk);
      bu.h[1] = *(const v8bf*)(Wrow + j * wstep + k0 + 16 + bk);
      v16bf b = bu.v;
      acc[j] = __builtin_amdgcn_wmma_f32_16x16x32_bf16(
          /*neg_a=*/false, a.v, /*neg_b=*/false, b,
          /*c_mod=*/(short)0, acc[j], /*reuse_a=*/false, /*reuse_b=*/false);
    }
  }

  const long long rbase = m0 + half * 8;  // C row = r + 8*half
#pragma unroll
  for (int j = 0; j < 4; ++j) {
    const int col = n0 + j * 16 + mrow;
    const float bb = bias ? bias[col] : 0.0f;
#pragma unroll
    for (int r = 0; r < 8; ++r) {
      float v = acc[j][r] + bb;
      if (relu) v = fmaxf(v, 0.0f);
      long long idx = (rbase + r) * (long long)ldc + col;
      if (Cf) Cf[idx] = v;
      if (Cb) Cb[idx] = (bf16_t)v;
    }
  }
}

// --------------------------- BatchNorm -------------------------------------

__global__ void bn_stats_kernel(const float* __restrict__ h,
                                float* __restrict__ sums,
                                float* __restrict__ sq,
                                int M, int D) {
  const int c = blockIdx.x;
  float s = 0.0f, q = 0.0f;
  for (int i = threadIdx.x; i < M; i += blockDim.x) {
    float v = h[(long long)i * D + c];
    s += v; q += v * v;
  }
  __shared__ float ss[256];
  __shared__ float qq[256];
  ss[threadIdx.x] = s; qq[threadIdx.x] = q;
  __syncthreads();
  for (int o = blockDim.x >> 1; o > 0; o >>= 1) {
    if ((int)threadIdx.x < o) {
      ss[threadIdx.x] += ss[threadIdx.x + o];
      qq[threadIdx.x] += qq[threadIdx.x + o];
    }
    __syncthreads();
  }
  if (threadIdx.x == 0) { sums[c] = ss[0]; sq[c] = qq[0]; }
}

__global__ void bn_apply_kernel(const float* __restrict__ h,
                                const float* __restrict__ sums,
                                const float* __restrict__ sq,
                                const float* __restrict__ g,
                                const float* __restrict__ bt,
                                float* __restrict__ xout,
                                long long M, int D, float invM) {
  long long n = M * (long long)D;
  long long t = (long long)blockIdx.x * blockDim.x + threadIdx.x;
  long long st = (long long)gridDim.x * blockDim.x;
  for (; t < n; t += st) {
    int c = (int)(t % D);
    float mean = sums[c] * invM;
    float var = sq[c] * invM - mean * mean;
    float v = (h[t] - mean) * rsqrtf(var + 1e-5f) * g[c] + bt[c];
    xout[t] = fmaxf(v, 0.0f);
  }
}

// --------------------------- mean pool -------------------------------------

__global__ void pool_accum_kernel(const float* __restrict__ x,
                                  const int* __restrict__ seg,
                                  float* __restrict__ pooled,
                                  float* __restrict__ cnt,
                                  long long M, int D) {
  long long n = M * (long long)D;
  long long t = (long long)blockIdx.x * blockDim.x + threadIdx.x;
  long long st = (long long)gridDim.x * blockDim.x;
  for (; t < n; t += st) {
    long long i = t / D;
    int f = (int)(t - i * D);
    int b = seg[i];
    atomicAdd(&pooled[(long long)b * D + f], x[t]);
    if (f == 0) atomicAdd(&cnt[b], 1.0f);
  }
}

__global__ void pool_div_kernel(float* __restrict__ pooled,
                                const float* __restrict__ cnt, int B, int D) {
  long long n = (long long)B * D;
  long long t = (long long)blockIdx.x * blockDim.x + threadIdx.x;
  long long st = (long long)gridDim.x * blockDim.x;
  for (; t < n; t += st) pooled[t] /= fmaxf(cnt[t / D], 1.0f);
}

// --------------------------- final matvec ----------------------------------

__global__ void final_matvec_kernel(const float* __restrict__ p1,
                                    const float* __restrict__ w,
                                    const float* __restrict__ b,
                                    float* __restrict__ out, int B, int K) {
  int i = blockIdx.x * blockDim.x + threadIdx.x;
  if (i >= B) return;
  float s = b[0];
  for (int k = 0; k < K; ++k) s += p1[(long long)i * K + k] * w[k];
  out[i] = s;
}

// --------------------------- host helpers ----------------------------------

static void launch_gemm(hipStream_t s, const bf16_t* A, int lda,
                        const bf16_t* W, int ldw, const float* bias,
                        float* Cf, bf16_t* Cb, int ldc,
                        int Mpad, int N, int Kpad, int relu) {
  dim3 g((N + 255) / 256, Mpad / 16);
  wmma_gemm_kernel<<<g, dim3(128), 0, s>>>(A, lda, W, ldw, bias, Cf, Cb, ldc,
                                           N, Kpad, relu);
}

static void launch_prep_wt(hipStream_t s, const float* W, bf16_t* Wt,
                           int K, int N, int Kpad) {
  long long n = (long long)N * Kpad;
  prep_wt_kernel<<<nblocks(n), TPB, 0, s>>>(W, Wt, K, N, Kpad);
}

// One GIN stack: 5 layers of scatter-add + MLP(WMMA) + BatchNorm.
static void run_gin_stack(hipStream_t s, const float* x0, int din, int M,
                          const int* src, const int* dst, long long E,
                          void* const* d_in, int pbase,
                          float* xbuf, float* aggbuf, bf16_t* Abuf,
                          bf16_t* Hbuf, bf16_t* Wt, float* bn) {
  const int Mpad = (M + 15) & ~15;
  const float* x = x0;
  int D = din;
  for (int l = 0; l < 5; ++l) {
    const float* W1 = (const float*)d_in[pbase + 6 * l + 0];
    const float* b1 = (const float*)d_in[pbase + 6 * l + 1];
    const float* W2 = (const float*)d_in[pbase + 6 * l + 2];
    const float* b2 = (const float*)d_in[pbase + 6 * l + 3];
    const float* g  = (const float*)d_in[pbase + 6 * l + 4];
    const float* bt = (const float*)d_in[pbase + 6 * l + 5];
    const int Kpad = (D + 31) & ~31;

    long long nAgg = (long long)M * D;
    fill_zero_f32<<<nblocks(nAgg), TPB, 0, s>>>(aggbuf, nAgg);
    scatter_add_kernel<<<nblocks(E * (long long)D), TPB, 0, s>>>(
        x, src, dst, aggbuf, E, D);
    prep_a_kernel<<<nblocks((long long)Mpad * Kpad), TPB, 0, s>>>(
        x, aggbuf, Abuf, M, Mpad, D, Kpad);

    // h1 = relu((x+agg) @ W1 + b1)   -> bf16
    launch_prep_wt(s, W1, Wt, D, HIDDEN, Kpad);
    launch_gemm(s, Abuf, Kpad, Wt, Kpad, b1, nullptr, Hbuf, HIDDEN,
                Mpad, HIDDEN, Kpad, 1);
    // h2 = relu(h1 @ W2 + b2)        -> f32 (reuse aggbuf)
    launch_prep_wt(s, W2, Wt, HIDDEN, HIDDEN, HIDDEN);
    launch_gemm(s, Hbuf, HIDDEN, Wt, HIDDEN, b2, aggbuf, nullptr, HIDDEN,
                Mpad, HIDDEN, HIDDEN, 1);

    // BatchNorm (training-mode batch stats) + ReLU -> xbuf
    fill_zero_f32<<<1, TPB, 0, s>>>(bn, 512);
    bn_stats_kernel<<<HIDDEN, 256, 0, s>>>(aggbuf, bn, bn + HIDDEN, M, HIDDEN);
    bn_apply_kernel<<<nblocks((long long)M * HIDDEN), TPB, 0, s>>>(
        aggbuf, bn, bn + HIDDEN, g, bt, xbuf, (long long)M, HIDDEN,
        1.0f / (float)M);
    x = xbuf;
    D = HIDDEN;
  }
}

// --------------------------- entry point -----------------------------------
//
// Input layout (flattened setup_inputs() dict order):
//  0 lig_x [NL,26] 1 prot_x [NP,20] 2 esm [B,1280] 3 lig_ei [2,EL]
//  4 prot_ei [2,EP] 5 lig_batch [NL] 6 prot_batch [NP]
//  7..36  lig_params  (5 x {W1,b1,W2,b2,g,bt})
// 37..66  prot_params (5 x {W1,b1,W2,b2,g,bt})
// 67 virtual_lig  68 virtual_prot   (dead: softmax over 1 key == 1)
// 69..76 attn_lig  {Wq,Wk,Wv,bq,bk,bv,Wo,bo}
// 77..84 attn_prot {Wq,Wk,Wv,bq,bk,bv,Wo,bo}
// 85..88 esm_proj {W1,b1,W2,b2}
// 89..92 fusion   {Wf1,bf1,Wf2,bf2}
// 93..96 predictor{Wp1,bp1,Wp2,bp2}

extern "C" void kernel_launch(void* const* d_in, const int* in_sizes, int n_in,
                              void* d_out, int out_size, void* d_ws,
                              size_t ws_size, hipStream_t stream) {
  (void)n_in; (void)out_size; (void)ws_size;

  const int NL = in_sizes[0] / 26;
  const int NP = in_sizes[1] / 20;
  const int B  = in_sizes[2] / 1280;
  const long long EL = in_sizes[3] / 2;
  const long long EP = in_sizes[4] / 2;
  const int NLp = (NL + 15) & ~15;
  const int NPp = (NP + 15) & ~15;
  const int Bp  = (B + 15) & ~15;
  const int MAXM = NPp > NLp ? NPp : NLp;

  const float* lig_x  = (const float*)d_in[0];
  const float* prot_x = (const float*)d_in[1];
  const float* esm    = (const float*)d_in[2];
  const int* lig_ei   = (const int*)d_in[3];
  const int* prot_ei  = (const int*)d_in[4];
  const int* lig_batch  = (const int*)d_in[5];
  const int* prot_batch = (const int*)d_in[6];

  // ---- workspace carve ----
  char* wsp = (char*)d_ws;
  size_t off = 0;
  auto alloc = [&](size_t bytes) -> void* {
    void* p = wsp + off;
    off = (off + bytes + 255) & ~(size_t)255;
    return p;
  };
  float*  xbuf   = (float*) alloc((size_t)MAXM * HIDDEN * 4);
  float*  aggbuf = (float*) alloc((size_t)MAXM * HIDDEN * 4); // also h2 (f32)
  bf16_t* Abuf   = (bf16_t*)alloc((size_t)MAXM * HIDDEN * 2);
  bf16_t* Hbuf   = (bf16_t*)alloc((size_t)MAXM * HIDDEN * 2);
  bf16_t* Wt     = (bf16_t*)alloc((size_t)512 * 1280 * 2);
  float*  bn     = (float*) alloc(512 * 4);
  float*  prot_pooled = (float*)alloc((size_t)Bp * HIDDEN * 4);
  float*  lig_pooled  = (float*)alloc((size_t)Bp * HIDDEN * 4);
  float*  cnt    = (float*) alloc((size_t)Bp * 4);
  bf16_t* hA     = (bf16_t*)alloc((size_t)Bp * 1280 * 2);
  bf16_t* hT1    = (bf16_t*)alloc((size_t)Bp * 512 * 2);
  bf16_t* hT2    = (bf16_t*)alloc((size_t)Bp * 256 * 2);
  bf16_t* hCat   = (bf16_t*)alloc((size_t)Bp * 768 * 2);
  float*  hP1    = (float*) alloc((size_t)Bp * 128 * 4);

  // ---- protein GIN stack + mean pool ----
  run_gin_stack(stream, prot_x, 20, NP, prot_ei, prot_ei + EP, EP, d_in, 37,
                xbuf, aggbuf, Abuf, Hbuf, Wt, bn);
  fill_zero_f32<<<nblocks((long long)Bp * HIDDEN), TPB, 0, stream>>>(
      prot_pooled, (long long)Bp * HIDDEN);
  fill_zero_f32<<<1, TPB, 0, stream>>>(cnt, Bp);
  pool_accum_kernel<<<nblocks((long long)NP * HIDDEN), TPB, 0, stream>>>(
      xbuf, prot_batch, prot_pooled, cnt, NP, HIDDEN);
  pool_div_kernel<<<nblocks((long long)B * HIDDEN), TPB, 0, stream>>>(
      prot_pooled, cnt, B, HIDDEN);

  // ---- ligand GIN stack + mean pool ----
  run_gin_stack(stream, lig_x, 26, NL, lig_ei, lig_ei + EL, EL, d_in, 7,
                xbuf, aggbuf, Abuf, Hbuf, Wt, bn);
  fill_zero_f32<<<nblocks((long long)Bp * HIDDEN), TPB, 0, stream>>>(
      lig_pooled, (long long)Bp * HIDDEN);
  fill_zero_f32<<<1, TPB, 0, stream>>>(cnt, Bp);
  pool_accum_kernel<<<nblocks((long long)NL * HIDDEN), TPB, 0, stream>>>(
      xbuf, lig_batch, lig_pooled, cnt, NL, HIDDEN);
  pool_div_kernel<<<nblocks((long long)B * HIDDEN), TPB, 0, stream>>>(
      lig_pooled, cnt, B, HIDDEN);

  // ---- lig_feat = (prot_pooled @ Wv_l + bv_l) @ Wo_l + bo_l -> hCat[:,0:256]
  prep_a_kernel<<<nblocks((long long)Bp * 256), TPB, 0, stream>>>(
      prot_pooled, nullptr, hA, B, Bp, 256, 256);
  launch_prep_wt(stream, (const float*)d_in[69 + 2], Wt, 256, 256, 256);
  launch_gemm(stream, hA, 256, Wt, 256, (const float*)d_in[69 + 5],
              nullptr, hT2, 256, Bp, 256, 256, 0);
  launch_prep_wt(stream, (const float*)d_in[69 + 6], Wt, 256, 256, 256);
  launch_gemm(stream, hT2, 256, Wt, 256, (const float*)d_in[69 + 7],
              nullptr, hCat + 0, 768, Bp, 256, 256, 0);

  // ---- prot_feat = (lig_pooled @ Wv_p + bv_p) @ Wo_p + bo_p -> hCat[:,256:512]
  prep_a_kernel<<<nblocks((long long)Bp * 256), TPB, 0, stream>>>(
      lig_pooled, nullptr, hA, B, Bp, 256, 256);
  launch_prep_wt(stream, (const float*)d_in[77 + 2], Wt, 256, 256, 256);
  launch_gemm(stream, hA, 256, Wt, 256, (const float*)d_in[77 + 5],
              nullptr, hT2, 256, Bp, 256, 256, 0);
  launch_prep_wt(stream, (const float*)d_in[77 + 6], Wt, 256, 256, 256);
  launch_gemm(stream, hT2, 256, Wt, 256, (const float*)d_in[77 + 7],
              nullptr, hCat + 256, 768, Bp, 256, 256, 0);

  // ---- esm_feat = relu(relu(esm @ W1 + b1) @ W2 + b2) -> hCat[:,512:768]
  prep_a_kernel<<<nblocks((long long)Bp * 1280), TPB, 0, stream>>>(
      esm, nullptr, hA, B, Bp, 1280, 1280);
  launch_prep_wt(stream, (const float*)d_in[85 + 0], Wt, 1280, 256, 1280);
  launch_gemm(stream, hA, 1280, Wt, 1280, (const float*)d_in[85 + 1],
              nullptr, hT2, 256, Bp, 256, 1280, 1);
  launch_prep_wt(stream, (const float*)d_in[85 + 2], Wt, 256, 256, 256);
  launch_gemm(stream, hT2, 256, Wt, 256, (const float*)d_in[85 + 3],
              nullptr, hCat + 512, 768, Bp, 256, 256, 1);

  // ---- fusion: relu(relu(concat @ Wf1 + bf1) @ Wf2 + bf2)
  launch_prep_wt(stream, (const float*)d_in[89 + 0], Wt, 768, 512, 768);
  launch_gemm(stream, hCat, 768, Wt, 768, (const float*)d_in[89 + 1],
              nullptr, hT1, 512, Bp, 512, 768, 1);
  launch_prep_wt(stream, (const float*)d_in[89 + 2], Wt, 512, 256, 512);
  launch_gemm(stream, hT1, 512, Wt, 512, (const float*)d_in[89 + 3],
              nullptr, hT2, 256, Bp, 256, 512, 1);

  // ---- predictor: relu(fused @ Wp1 + bp1) @ Wp2 + bp2 -> [B,1]
  launch_prep_wt(stream, (const float*)d_in[93 + 0], Wt, 256, 128, 256);
  launch_gemm(stream, hT2, 256, Wt, 256, (const float*)d_in[93 + 1],
              hP1, nullptr, 128, Bp, 128, 256, 1);
  final_matvec_kernel<<<(B + 127) / 128, 128, 0, stream>>>(
      hP1, (const float*)d_in[93 + 2], (const float*)d_in[93 + 3],
      (float*)d_out, B, 128);
}
